// TimeGradPredictionNetwork_76398878261699
// MI455X (gfx1250) — matrix operand order repaired and checked by reading, not compiled, not run on hardware
//
#include <hip/hip_runtime.h>
#include <hip/hip_bf16.h>
#include <math.h>

typedef __attribute__((ext_vector_type(16))) _Float16 v16h;
typedef __attribute__((ext_vector_type(8)))  _Float16 v8h;
typedef __attribute__((ext_vector_type(8)))  float    v8f;

#define T_LEN 4096
#define F_IN  132
#define D_DIM 128
#define H_DIM 100
#define RC    8
#define RL    8
#define RH    64
#define STEPS 100

// ---------------- workspace layout ----------------
#define WS_XG0    0              // 4096*400
#define WS_RNN    1638400        // 4096*100
#define WS_C1     2048000        // 4096*64
#define WS_CC     2310144        // 4096*128
#define WS_SCALE  2834432        // 128
#define WS_TAB    2834560        // 5*100 packed at +0,+100,+200,+300,+400 (640 reserved)
#define WS_DSTEP  2835200        // 8*100*8
#define WS_B0SUM  2841600        // 400
#define WS_B1SUM  2842000        // 400
#define WS_FLOAT_END 2842400
#define HB        (2*WS_FLOAT_END)
#define WH_PKTF   (HB)                      // 4096*160
#define WH_PKW0   (WH_PKTF + 4096*160)      // 400*160
#define WH_PKRNN  (WH_PKW0 + 400*160)       // 4096*128
#define WH_PKCU1  (WH_PKRNN + 4096*128)     // 64*128
#define WH_PKC1   (WH_PKCU1 + 64*128)       // 4096*64
#define WH_PKCU2  (WH_PKC1 + 4096*64)       // 128*64

__device__ __forceinline__ float sigf(float x){ return 1.0f/(1.0f+__expf(-x)); }
__device__ __forceinline__ float leakyf(float x){ return x>0.0f ? x : 0.4f*x; }

__device__ __forceinline__ unsigned hash32(unsigned x){
  x ^= x>>16; x *= 0x7feb352du; x ^= x>>15; x *= 0x846ca68bu; x ^= x>>16; return x;
}
__device__ __forceinline__ float nrand(unsigned s){
  unsigned a = hash32(s*0x9E3779B9u + 0x85ebca6bu);
  unsigned b = hash32(a ^ 0xdeadbeefu);
  float u1 = (float)(a>>8)*(1.0f/16777216.0f) + 1e-7f;
  float u2 = (float)(b>>8)*(1.0f/16777216.0f);
  return sqrtf(-2.0f*__logf(u1))*__cosf(6.2831853f*u2);
}

// ---- CDNA5 async global->LDS copy (16B per active lane), ASYNCcnt tracked ----
__device__ __forceinline__ void async_ld_b128(void* lds, const void* g){
  unsigned l = (unsigned)(size_t)lds;        // low 32 bits of generic LDS ptr == LDS offset
  asm volatile("global_load_async_to_lds_b128 %0, %1, off"
               :: "v"(l), "v"(g) : "memory");
}
__device__ __forceinline__ void async_wait(){
  asm volatile("s_wait_asynccnt 0x0" ::: "memory");
}

// ---------------- K1: posterior tables, diffusion embedding, bias sums ----
__global__ void tables_kernel(const float* __restrict__ de1w, const float* __restrict__ de1b,
                              const float* __restrict__ de2w, const float* __restrict__ de2b,
                              const float* __restrict__ dpw,  const float* __restrict__ dpb,
                              const float* __restrict__ bih0, const float* __restrict__ bhh0,
                              const float* __restrict__ bih1, const float* __restrict__ bhh1,
                              float* __restrict__ ws){
  int tid = threadIdx.x;
  if (tid == 0){
    double ac = 1.0, acp_prev = 1.0;
    for (int t = 0; t < STEPS; ++t){
      double beta  = 0.0001 + (0.1 - 0.0001) * (double)t / (double)(STEPS-1);
      double alpha = 1.0 - beta;
      double acp   = acp_prev;
      ac *= alpha;
      acp_prev = ac;
      ws[WS_TAB +       t] = (float)sqrt(1.0/ac);
      ws[WS_TAB + 100 + t] = (float)sqrt(1.0/ac - 1.0);
      ws[WS_TAB + 200 + t] = (float)(beta * sqrt(acp) / (1.0 - ac));
      ws[WS_TAB + 300 + t] = (float)((1.0 - acp) * sqrt(alpha) / (1.0 - ac));
      double pv = beta * (1.0 - acp) / (1.0 - ac);
      ws[WS_TAB + 400 + t] = (float)log(pv > 1e-20 ? pv : 1e-20);
    }
    for (int t = 500; t < 512; ++t) ws[WS_TAB + t] = 0.0f;  // pad for async stage
  }
  if (tid < 400){
    ws[WS_B0SUM + tid] = bih0[tid] + bhh0[tid];
    ws[WS_B1SUM + tid] = bih1[tid] + bhh1[tid];
  }
  if (tid < STEPS){
    int t = tid;
    float dt[32];
    #pragma unroll
    for (int d = 0; d < 16; ++d){
      float arg = (float)t * __powf(10.0f, (float)d * 0.25f);
      dt[d]    = __sinf(arg);
      dt[16+d] = __cosf(arg);
    }
    float e1[RH];
    for (int u = 0; u < RH; ++u){
      float s = de1b[u];
      for (int j = 0; j < 32; ++j) s += dt[j]*de1w[u*32+j];
      e1[u] = s * sigf(s);
    }
    float e2[RH];
    for (int u = 0; u < RH; ++u){
      float s = de2b[u];
      for (int j = 0; j < RH; ++j) s += e1[j]*de2w[u*RH+j];
      e2[u] = s * sigf(s);
    }
    for (int i = 0; i < RL; ++i)
      for (int c = 0; c < RC; ++c){
        float s = dpb[i*RC + c];
        const float* w = &dpw[(i*RC + c)*RH];
        for (int u = 0; u < RH; ++u) s += e2[u]*w[u];
        ws[WS_DSTEP + (i*STEPS + t)*RC + c] = s;
      }
  }
}

// ---------------- K0: scale[d] = max(mean_t |tf[t,d]|, 1e-10) --------------
__global__ void scale_kernel(const float* __restrict__ tf, float* __restrict__ ws){
  __shared__ float red[256];
  int d = blockIdx.x;
  float s = 0.0f;
  for (int t = threadIdx.x; t < T_LEN; t += 256) s += fabsf(tf[t*F_IN + d]);
  red[threadIdx.x] = s; __syncthreads();
  for (int o = 128; o > 0; o >>= 1){
    if (threadIdx.x < o) red[threadIdx.x] += red[threadIdx.x + o];
    __syncthreads();
  }
  if (threadIdx.x == 0) ws[WS_SCALE + d] = fmaxf(red[0]/(float)T_LEN, 1e-10f);
}

// ---------------- pack: f32[rows x K] -> f16[rows x Kp], zero pad ----------
__global__ void pack_f16(const float* __restrict__ src, _Float16* __restrict__ dst,
                         int rows, int K, int Kp){
  int idx = blockIdx.x*256 + threadIdx.x;
  if (idx >= rows*Kp) return;
  int r = idx / Kp, k = idx - r*Kp;
  dst[idx] = (_Float16)((k < K) ? src[(size_t)r*K + k] : 0.0f);
}

// ---------------- WMMA GEMM on packed f16: C = act(A @ W^T + bias) ---------
__global__ void gemm_nt_wmma(const _Float16* __restrict__ A, const _Float16* __restrict__ W,
                             const float* __restrict__ bias, float* __restrict__ C,
                             int N, int Kp, int act){
  const int lane = threadIdx.x;
  const int half = lane >> 4;
  const int l16  = lane & 15;
  const _Float16* aPtr = A + (size_t)(blockIdx.x*16 + l16)*Kp;
  const _Float16* wPtr = W + (size_t)(blockIdx.y*16 + l16)*Kp;
  v8f acc = {};
  for (int k0 = 0; k0 < Kp; k0 += 32){
    const _Float16* ap = aPtr + k0 + half*16;  // 32B aligned
    const _Float16* bp = wPtr + k0 + half*16;  // 32B aligned
    __builtin_prefetch(ap + 128, 0, 1);
    __builtin_prefetch(bp + 128, 0, 1);
    v8h a_lo = *(const v8h*)(ap);
    v8h a_hi = *(const v8h*)(ap + 8);
    v8h b_lo = *(const v8h*)(bp);
    v8h b_hi = *(const v8h*)(bp + 8);
    v16h af = __builtin_shufflevector(a_lo, a_hi, 0,1,2,3,4,5,6,7,8,9,10,11,12,13,14,15);
    v16h bf = __builtin_shufflevector(b_lo, b_hi, 0,1,2,3,4,5,6,7,8,9,10,11,12,13,14,15);
    acc = __builtin_amdgcn_wmma_f32_16x16x32_f16(false, af, false, bf, (short)0, acc, false, false);
  }
  const int n = blockIdx.y*16 + l16;
  const float bv = bias ? bias[n] : 0.0f;
  #pragma unroll
  for (int r = 0; r < 8; ++r){
    int m = blockIdx.x*16 + half*8 + r;
    float v = acc[r] + bv;
    if (act == 1) v = leakyf(v);
    C[(size_t)m*N + n] = v;
  }
}

// ---------------- K3: persistent sequential LSTM (2 layers x 2 passes) ----
__global__ void lstm_kernel(const float* __restrict__ xg0,
                            const float* __restrict__ Whh0,
                            const float* __restrict__ Wih1,
                            const float* __restrict__ Whh1,
                            const float* __restrict__ b1sum,
                            float* __restrict__ rnn){
  __shared__ float h0[H_DIM], c0[H_DIM], h1[H_DIM], c1[H_DIM];
  __shared__ float g0[400], g1[400];
  int tid = threadIdx.x;
  if (tid < H_DIM){ h0[tid]=0.f; c0[tid]=0.f; h1[tid]=0.f; c1[tid]=0.f; }
  __syncthreads();
  for (int pass = 0; pass < 2; ++pass){
    for (int t = 0; t < T_LEN; ++t){
      if (tid < 400){
        float a = xg0[(size_t)t*400 + tid];
        const float4* w4 = (const float4*)&Whh0[tid*H_DIM];
        #pragma unroll 5
        for (int k = 0; k < 25; ++k){
          float4 wv = w4[k];
          a += wv.x*h0[k*4] + wv.y*h0[k*4+1] + wv.z*h0[k*4+2] + wv.w*h0[k*4+3];
        }
        g0[tid] = a;
      }
      __syncthreads();
      if (tid < H_DIM){
        float ii = sigf(g0[tid]), ff = sigf(g0[100+tid]);
        float gg = tanhf(g0[200+tid]), oo = sigf(g0[300+tid]);
        float cn = ff*c0[tid] + ii*gg;
        c0[tid] = cn; h0[tid] = oo*tanhf(cn);
      }
      __syncthreads();
      if (tid < 400){
        float a = b1sum[tid];
        const float4* wi4 = (const float4*)&Wih1[tid*H_DIM];
        const float4* wh4 = (const float4*)&Whh1[tid*H_DIM];
        #pragma unroll 5
        for (int k = 0; k < 25; ++k){
          float4 wi = wi4[k], wh = wh4[k];
          a += wi.x*h0[k*4] + wi.y*h0[k*4+1] + wi.z*h0[k*4+2] + wi.w*h0[k*4+3];
          a += wh.x*h1[k*4] + wh.y*h1[k*4+1] + wh.z*h1[k*4+2] + wh.w*h1[k*4+3];
        }
        g1[tid] = a;
      }
      __syncthreads();
      if (tid < H_DIM){
        float ii = sigf(g1[tid]), ff = sigf(g1[100+tid]);
        float gg = tanhf(g1[200+tid]), oo = sigf(g1[300+tid]);
        float cn = ff*c1[tid] + ii*gg;
        c1[tid] = cn;
        float hn = oo*tanhf(cn);
        h1[tid] = hn;
        if (pass == 1) rnn[(size_t)t*H_DIM + tid] = hn;
      }
      __syncthreads();
    }
  }
}

// ---------------- K5: persistent diffusion sampler, one wave per row -------
__global__ void diffusion_kernel(const float* __restrict__ ws,
                                 const float* __restrict__ inpw, const float* __restrict__ inpb,
                                 const float* __restrict__ dcw,  const float* __restrict__ dcb,
                                 const float* __restrict__ cpw,  const float* __restrict__ cpb,
                                 const float* __restrict__ opw,  const float* __restrict__ opb,
                                 const float* __restrict__ skw,  const float* __restrict__ skb,
                                 const float* __restrict__ outw, const float* __restrict__ outb,
                                 float* __restrict__ out){
  __shared__ float    sh_h [RC][132];
  __shared__ float    sh_hp[RC][132];
  __shared__ float    sh_y [16][144];
  __shared__ float    sh_sk[RC][132];
  __shared__ float    sh_x [D_DIM];
  __shared__ float    sh_cp[144];
  __shared__ float    sh_eps[D_DIM];
  __shared__ _Float16 sh_wdc[RL][16][32];
  __shared__ float    sh_opw[RL][16][RC];
  __shared__ float    sh_bsum[RL][16];
  __shared__ float    sh_cpw[RL][16];
  __shared__ float    sh_opb[RL][16];
  __shared__ float    sh_skw[192];
  __shared__ float    sh_outw[24];
  __shared__ float    sh_misc[32];
  __shared__ __align__(16) float sh_tab[512];     // async-staged posterior tables
  __shared__ __align__(16) float sh_ccrow[128];   // async-staged cond row
  __shared__ __align__(16) float sh_dst[RC][RC];  // async-staged per-step dstep slice

  const int lane = threadIdx.x;
  const int row  = blockIdx.x;
  const int half = lane >> 4;
  const int l16  = lane & 15;

  const float* cc    = ws + WS_CC + (size_t)row*D_DIM;
  const float* gtab  = ws + WS_TAB;
  const float* dstep = ws + WS_DSTEP;
  const float* scale = ws + WS_SCALE;

  // ---- issue async DMA of per-block constants (overlaps LDS weight staging) ----
  #pragma unroll
  for (int it = 0; it < 4; ++it)
    async_ld_b128(&sh_tab[it*128 + lane*4], gtab + it*128 + lane*4);
  async_ld_b128(&sh_ccrow[lane*4], cc + lane*4);

  // ---- stage constants into LDS (single wave: DS ops are in-order) ----
  for (int idx = lane; idx < RL*16*32; idx += 32){
    int i  = idx >> 9;
    int oc = (idx >> 5) & 15;
    int k  = idx & 31;
    float v = (k < 24) ? dcw[((i*16 + oc)*RC + k/3)*3 + (k%3)] : 0.0f;
    sh_wdc[i][oc][k] = (_Float16)v;
  }
  for (int idx = lane; idx < RL*16*RC; idx += 32){
    int i = idx >> 7; int oc = (idx >> 3) & 15; int ic = idx & 7;
    sh_opw[i][oc][ic] = opw[(i*16 + oc)*RC + ic];
  }
  for (int idx = lane; idx < RL*16; idx += 32){
    int i = idx >> 4, oc = idx & 15;
    sh_bsum[i][oc] = dcb[idx] + cpb[idx];
    sh_cpw[i][oc]  = cpw[idx];
    sh_opb[i][oc]  = opb[idx];
  }
  for (int idx = lane; idx < 192; idx += 32) sh_skw[idx] = skw[idx];
  if (lane < 24) sh_outw[lane] = outw[lane];
  if (lane < 8){ sh_misc[lane] = skb[lane]; sh_misc[8+lane] = inpw[lane]; sh_misc[16+lane] = inpb[lane]; }
  if (lane == 0) sh_misc[24] = outb[0];

  // ---- init x0 ~ N(0,1) ----
  for (int d = lane; d < D_DIM; d += 32) sh_x[d] = nrand((unsigned)(row*977u + d));

  async_wait();   // tables + cond row resident
  for (int j = lane; j < 144; j += 32)
    sh_cp[j] = (j < 132) ? sh_ccrow[(j + 126) & 127] : 0.0f;

  for (int t = STEPS-1; t >= 0; --t){
    // issue async gather of this step's dstep slice (8 RB x 8 ch, 16B chunks)
    {
      int c  = lane & 15;             // lanes 16..31 duplicate (same src, same dst)
      int i  = c >> 1, hh = (c & 1)*4;
      async_ld_b128(&sh_dst[i][hh], dstep + ((size_t)i*STEPS + t)*RC + hh);
    }

    // input conv (k=1 over circ_pad(x,2)) + zero skips  (hides async latency)
    for (int j = lane; j < 132; j += 32){
      float xp = sh_x[(j + 126) & 127];
      #pragma unroll
      for (int ic = 0; ic < RC; ++ic){
        sh_h[ic][j]  = leakyf(sh_misc[8+ic]*xp + sh_misc[16+ic]);
        sh_sk[ic][j] = 0.0f;
      }
    }
    async_wait();

    for (int i = 0; i < RL; ++i){
      const int dil = (i & 1) ? 2 : 1;

      // hp = h + dstep (dstep now in LDS, b128 reads)
      {
        float4 d03 = *(const float4*)&sh_dst[i][0];
        float4 d47 = *(const float4*)&sh_dst[i][4];
        for (int j = lane; j < 132; j += 32){
          sh_hp[0][j] = sh_h[0][j] + d03.x; sh_hp[1][j] = sh_h[1][j] + d03.y;
          sh_hp[2][j] = sh_h[2][j] + d03.z; sh_hp[3][j] = sh_h[3][j] + d03.w;
          sh_hp[4][j] = sh_h[4][j] + d47.x; sh_hp[5][j] = sh_h[5][j] + d47.y;
          sh_hp[6][j] = sh_h[6][j] + d47.z; sh_hp[7][j] = sh_h[7][j] + d47.w;
        }
      }

      // A-fragment from LDS (pre-padded, unconditional 16B loads)
      v16h af;
      {
        const _Float16* wrow = &sh_wdc[i][l16][half*16];
        v8h lo = *(const v8h*)(wrow);
        v8h hi = *(const v8h*)(wrow + 8);
        af = __builtin_shufflevector(lo, hi, 0,1,2,3,4,5,6,7,8,9,10,11,12,13,14,15);
      }

      // dilated conv as implicit GEMM: 9 N-tiles of 16 positions
      for (int nt = 0; nt < 9; ++nt){
        const int n  = nt*16 + l16;
        const int np = (n < 132) ? n : n - 132;
        const int kb = half*16;
        v16h bf;
        #pragma unroll
        for (int j = 0; j < 16; ++j){
          int k  = kb + j;
          int kc = (k < 24) ? k : 0;
          int ic = kc/3, kk = kc - ic*3;
          int src = np + (kk-1)*dil;
          src += (src < 0)    ? 132 : 0;
          src -= (src >= 132) ? 132 : 0;
          bf[j] = (_Float16)sh_hp[ic][src];
        }
        v8f acc = {};
        acc = __builtin_amdgcn_wmma_f32_16x16x32_f16(false, af, false, bf, (short)0, acc, false, false);
        const float cpv = sh_cp[n];
        #pragma unroll
        for (int r = 0; r < 8; ++r){
          int oc = half*8 + r;
          sh_y[oc][n] = acc[r] + sh_cpw[i][oc]*cpv + sh_bsum[i][oc];
        }
      }

      // gated activation + 1x1 conv + residual/skip (VALU)
      for (int j = lane; j < 132; j += 32){
        float z[RC];
        #pragma unroll
        for (int ic = 0; ic < RC; ++ic)
          z[ic] = sigf(sh_y[ic][j]) * tanhf(sh_y[RC+ic][j]);
        #pragma unroll
        for (int oc = 0; oc < 16; ++oc){
          float a = sh_opb[i][oc];
          #pragma unroll
          for (int ic = 0; ic < RC; ++ic) a += sh_opw[i][oc][ic]*z[ic];
          a = leakyf(a);
          if (oc < RC) sh_h[oc][j]      = (sh_h[oc][j] + a)*0.70710678f;
          else         sh_sk[oc-RC][j] += a;
        }
      }
    }

    // skip conv (VALID k=3): 132 -> 130
    for (int j = lane; j < 130; j += 32){
      #pragma unroll
      for (int oc = 0; oc < RC; ++oc){
        float a = sh_misc[oc];
        #pragma unroll
        for (int ic = 0; ic < RC; ++ic)
          #pragma unroll
          for (int kk = 0; kk < 3; ++kk)
            a += sh_skw[(oc*RC + ic)*3 + kk] * (sh_sk[ic][j+kk]*0.35355339f);
        sh_y[oc][j] = leakyf(a);
      }
    }
    // out conv (VALID k=3): 130 -> 128
    for (int j = lane; j < D_DIM; j += 32){
      float a = sh_misc[24];
      #pragma unroll
      for (int ic = 0; ic < RC; ++ic)
        #pragma unroll
        for (int kk = 0; kk < 3; ++kk)
          a += sh_outw[ic*3 + kk]*sh_y[ic][j+kk];
      sh_eps[j] = a;
    }

    // posterior update (tables from LDS)
    const float sra = sh_tab[t], srm1 = sh_tab[100+t];
    const float pc1 = sh_tab[200+t], pc2 = sh_tab[300+t], plv = sh_tab[400+t];
    const float sg  = (t > 0) ? __expf(0.5f*plv) : 0.0f;
    for (int d = lane; d < D_DIM; d += 32){
      float xv = sh_x[d];
      float xr = sra*xv - srm1*sh_eps[d];
      float mn = pc1*xr + pc2*xv;
      sh_x[d] = mn + sg*nrand((unsigned)(row*131071u + (unsigned)t*524287u + d));
    }
  }

  for (int d = lane; d < D_DIM; d += 32)
    out[(size_t)row*D_DIM + d] = sh_x[d]*scale[d];
}

// ---------------------------------------------------------------------------
extern "C" void kernel_launch(void* const* d_in, const int* in_sizes, int n_in,
                              void* d_out, int out_size, void* d_ws, size_t ws_size,
                              hipStream_t stream){
  const float* time_feat = (const float*)d_in[0];
  const float* Wih0 = (const float*)d_in[1];
  const float* Whh0 = (const float*)d_in[2];
  const float* bih0 = (const float*)d_in[3];
  const float* bhh0 = (const float*)d_in[4];
  const float* Wih1 = (const float*)d_in[5];
  const float* Whh1 = (const float*)d_in[6];
  const float* bih1 = (const float*)d_in[7];
  const float* bhh1 = (const float*)d_in[8];
  const float* inpw = (const float*)d_in[9];
  const float* inpb = (const float*)d_in[10];
  const float* de1w = (const float*)d_in[11];
  const float* de1b = (const float*)d_in[12];
  const float* de2w = (const float*)d_in[13];
  const float* de2b = (const float*)d_in[14];
  const float* cu1w = (const float*)d_in[15];
  const float* cu1b = (const float*)d_in[16];
  const float* cu2w = (const float*)d_in[17];
  const float* cu2b = (const float*)d_in[18];
  const float* dcw  = (const float*)d_in[19];
  const float* dcb  = (const float*)d_in[20];
  const float* dpw  = (const float*)d_in[21];
  const float* dpb  = (const float*)d_in[22];
  const float* cpw  = (const float*)d_in[23];
  const float* cpb  = (const float*)d_in[24];
  const float* opw  = (const float*)d_in[25];
  const float* opb  = (const float*)d_in[26];
  const float* skw  = (const float*)d_in[27];
  const float* skb  = (const float*)d_in[28];
  const float* outw = (const float*)d_in[29];
  const float* outb = (const float*)d_in[30];

  float*     ws  = (float*)d_ws;
  _Float16*  wsh = (_Float16*)d_ws;
  float*     out = (float*)d_out;

  tables_kernel<<<1, 512, 0, stream>>>(de1w, de1b, de2w, de2b, dpw, dpb,
                                       bih0, bhh0, bih1, bhh1, ws);
  scale_kernel<<<D_DIM, 256, 0, stream>>>(time_feat, ws);
  pack_f16<<<(T_LEN*160 + 255)/256, 256, 0, stream>>>(time_feat, wsh + WH_PKTF, T_LEN, F_IN, 160);
  pack_f16<<<(400*160   + 255)/256, 256, 0, stream>>>(Wih0,      wsh + WH_PKW0,  400,  F_IN, 160);
  pack_f16<<<(64*128    + 255)/256, 256, 0, stream>>>(cu1w,      wsh + WH_PKCU1, 64,   H_DIM,128);
  pack_f16<<<(128*64    + 255)/256, 256, 0, stream>>>(cu2w,      wsh + WH_PKCU2, 128,  64,   64);
  gemm_nt_wmma<<<dim3(T_LEN/16, 400/16), 32, 0, stream>>>(
      wsh + WH_PKTF, wsh + WH_PKW0, ws + WS_B0SUM, ws + WS_XG0, 400, 160, 0);
  lstm_kernel<<<1, 512, 0, stream>>>(ws + WS_XG0, Whh0, Wih1, Whh1,
                                     ws + WS_B1SUM, ws + WS_RNN);
  pack_f16<<<(T_LEN*128 + 255)/256, 256, 0, stream>>>(ws + WS_RNN, wsh + WH_PKRNN, T_LEN, H_DIM, 128);
  gemm_nt_wmma<<<dim3(T_LEN/16, 64/16), 32, 0, stream>>>(
      wsh + WH_PKRNN, wsh + WH_PKCU1, cu1b, ws + WS_C1, 64, 128, 1);
  pack_f16<<<(T_LEN*64 + 255)/256, 256, 0, stream>>>(ws + WS_C1, wsh + WH_PKC1, T_LEN, 64, 64);
  gemm_nt_wmma<<<dim3(T_LEN/16, 128/16), 32, 0, stream>>>(
      wsh + WH_PKC1, wsh + WH_PKCU2, cu2b, ws + WS_CC, 128, 64, 1);
  diffusion_kernel<<<T_LEN, 32, 0, stream>>>(ws, inpw, inpb, dcw, dcb, cpw, cpb,
                                             opw, opb, skw, skb, outw, outb, out);
}
